// MultiHeadAttention_9371618640155
// MI455X (gfx1250) — compile-verified, matrix operand order checked by reading, not run on hardware
//
#include <hip/hip_runtime.h>

#define D_MODEL 1024
#define NH 16
#define HEAD 64
#define BATCH 2
#define SEQ 2048
#define ROWS (BATCH * SEQ)   // 4096

#define TILE_M 64
#define TILE_N 64
#define TILE_K 32
#define GEMM_THREADS 128
#define KV_TILE 64
#define ATT_THREADS 128

typedef __bf16 bf16;
typedef __attribute__((ext_vector_type(16))) __bf16 v16bf;
typedef __attribute__((ext_vector_type(8)))  float  v8f;
typedef __attribute__((ext_vector_type(4)))  int    v4i;

#ifndef __has_builtin
#define __has_builtin(x) 0
#endif
#if __has_builtin(__builtin_amdgcn_global_load_async_to_lds_b128)
#define HAVE_ASYNC_LDS 1
#else
#define HAVE_ASYNC_LDS 0
#endif

// ---- CDNA5 helpers ----------------------------------------------------------

__device__ __forceinline__ v8f wmma_bf16(v16bf a, v16bf b, v8f c) {
  // (neg_a, A, neg_b, B, c_mod, C, reuse_a, reuse_b)
  return __builtin_amdgcn_wmma_f32_16x16x32_bf16(false, a, false, b, (short)0, c,
                                                 false, false);
}

#if HAVE_ASYNC_LDS
__device__ __forceinline__ void async_copy16(const void* g, void* l) {
  // Signature (from clang diagnostic): (v4i AS1*, v4i AS3*, imm offset, imm cpol)
  __builtin_amdgcn_global_load_async_to_lds_b128(
      (__attribute__((address_space(1))) v4i*)g,
      (__attribute__((address_space(3))) v4i*)l, 0, 0);
}
__device__ __forceinline__ void wait_async0() {
#if __has_builtin(__builtin_amdgcn_s_wait_asynccnt)
  __builtin_amdgcn_s_wait_asynccnt(0);
#else
  asm volatile("s_wait_asynccnt 0x0" ::: "memory");
#endif
}
#endif

// K index for element j of a 16-bit A/B fragment (wave32 layout):
// lanes 0-15: K = j (j<8) or 16+(j-8); lanes 16-31: +8.
__device__ __forceinline__ int kmap(int lane, int j) {
  int base = (lane & 16) ? 8 : 0;
  return base + (j & 7) + ((j & 8) << 1);
}

// A-fragment (16 rows x 32 k) from row-major [16][ld] at column k0.
// Also serves as B-fragment when B must be read transposed (B[k][n] = src[n][k]).
__device__ __forceinline__ v16bf load_frag_rm(const bf16* src, int ld, int k0,
                                              int lane) {
  v16bf f;
  int r = lane & 15;
  #pragma unroll
  for (int j = 0; j < 16; ++j) f[j] = src[r * ld + k0 + kmap(lane, j)];
  return f;
}

// B-fragment from natural row-major [K][N] storage: B[k][n] = src[k*ld + n].
__device__ __forceinline__ v16bf load_frag_kn(const bf16* src, int ld, int k0,
                                              int n0, int lane) {
  v16bf f;
  int n = lane & 15;
  #pragma unroll
  for (int j = 0; j < 16; ++j) f[j] = src[(k0 + kmap(lane, j)) * ld + n0 + n];
  return f;
}

// 4x fp32 -> 4x bf16 packed 8-byte store (LDS).
__device__ __forceinline__ void store4_bf16(bf16* dst, float4 v) {
  union { bf16 h[4]; uint2 u; } t;
  t.h[0] = (bf16)v.x; t.h[1] = (bf16)v.y; t.h[2] = (bf16)v.z; t.h[3] = (bf16)v.w;
  *(uint2*)dst = t.u;
}

// ---- Kernel 1: weight fp32 -> bf16 -----------------------------------------

__global__ void pack_w_kernel(const float* __restrict__ src,
                              bf16* __restrict__ dst, int n) {
  int i = (blockIdx.x * blockDim.x + threadIdx.x) * 4;
  if (i < n) {
    float4 v = *(const float4*)&src[i];
    store4_bf16(&dst[i], v);
  }
}

// ---- Kernel 2: fused QKV projection (z = 0/1/2 -> Q/K/V) -------------------
// q = x @ Wq^T + bq ; outputs stored bf16 head-major [B][H][S][HEAD].

__global__ __launch_bounds__(GEMM_THREADS)
void qkv_proj_kernel(const float* __restrict__ x, const float* __restrict__ y,
                     const bf16* __restrict__ WqB, const bf16* __restrict__ WkB,
                     const bf16* __restrict__ WvB,
                     const float* __restrict__ bqv, const float* __restrict__ bkv,
                     const float* __restrict__ bvv,
                     bf16* __restrict__ Q, bf16* __restrict__ K,
                     bf16* __restrict__ V) {
  __shared__ __align__(16) bf16 As[TILE_M][TILE_K];
  __shared__ __align__(16) bf16 Bs[TILE_N][TILE_K];
  __shared__ __align__(16) bf16 Cs[TILE_M][TILE_N];

  const int which = blockIdx.z;  // 0=Q,1=K,2=V
  const float* src  = (which == 0) ? x : y;
  const bf16*  W    = (which == 0) ? WqB : (which == 1) ? WkB : WvB;
  const float* bias = (which == 0) ? bqv : (which == 1) ? bkv : bvv;
  bf16*        dst  = (which == 0) ? Q : (which == 1) ? K : V;

  const int m0 = blockIdx.x * TILE_M;
  const int n0 = blockIdx.y * TILE_N;
  const int tid = threadIdx.x;
  const int lane = tid & 31;
  const int wave = tid >> 5;
  const int mOff = (wave >> 1) * 32;
  const int nOff = (wave & 1) * 32;

  v8f c[2][2] = {};

  for (int k0 = 0; k0 < D_MODEL; k0 += TILE_K) {
    // A tile: 64x32 fp32 -> bf16. 512 float4 units / 128 threads.
    #pragma unroll
    for (int i = 0; i < 4; ++i) {
      int f4 = tid + i * GEMM_THREADS;      // 0..511
      int r = f4 >> 3, cc = (f4 & 7) * 4;
      float4 v = *(const float4*)&src[(size_t)(m0 + r) * D_MODEL + k0 + cc];
      store4_bf16(&As[r][cc], v);
    }
    // B tile: 64x32 bf16 = 256 x 16B units / 128 threads (async -> LDS).
    #pragma unroll
    for (int i = 0; i < 2; ++i) {
      int u = tid + i * GEMM_THREADS;       // 0..255
      int r = u >> 2, seg = (u & 3) * 8;
      const bf16* g = &W[(size_t)(n0 + r) * D_MODEL + k0 + seg];
#if HAVE_ASYNC_LDS
      async_copy16(g, &Bs[r][seg]);
#else
      *(uint4*)&Bs[r][seg] = *(const uint4*)g;
#endif
    }
#if HAVE_ASYNC_LDS
    wait_async0();
#endif
    __syncthreads();
    #pragma unroll
    for (int mi = 0; mi < 2; ++mi) {
      v16bf a = load_frag_rm(&As[mOff + mi * 16][0], TILE_K, 0, lane);
      #pragma unroll
      for (int ni = 0; ni < 2; ++ni) {
        v16bf b = load_frag_rm(&Bs[nOff + ni * 16][0], TILE_K, 0, lane);
        c[mi][ni] = wmma_bf16(a, b, c[mi][ni]);
      }
    }
    __syncthreads();
  }

  // Epilogue: +bias, bf16, stage in LDS, coalesced b128 stores.
  const int half = (lane >> 4) & 1;
  const int nl = lane & 15;
  #pragma unroll
  for (int mi = 0; mi < 2; ++mi)
    #pragma unroll
    for (int ni = 0; ni < 2; ++ni) {
      float bv = bias[n0 + nOff + ni * 16 + nl];
      #pragma unroll
      for (int r = 0; r < 8; ++r)
        Cs[mOff + mi * 16 + r + half * 8][nOff + ni * 16 + nl] =
            (bf16)(c[mi][ni][r] + bv);
    }
  __syncthreads();
  const int h = n0 >> 6;  // TILE_N == HEAD: one head per column tile
  #pragma unroll
  for (int i = 0; i < 4; ++i) {
    int u = tid + i * GEMM_THREADS;   // 0..511 (64 rows x 8 x uint4)
    int r = u >> 3, c8 = (u & 7) * 8;
    int m = m0 + r;
    int bb = m / SEQ, s = m % SEQ;
    *(uint4*)&dst[(((size_t)bb * NH + h) * SEQ + s) * HEAD + c8] =
        *(const uint4*)&Cs[r][c8];
  }
}

// ---- Kernel 3: flash attention per (b, h, 64-q-row block) ------------------

__global__ __launch_bounds__(ATT_THREADS)
void attention_kernel(const bf16* __restrict__ Q, const bf16* __restrict__ Kt,
                      const bf16* __restrict__ Vt, const float* __restrict__ mask,
                      bf16* __restrict__ ctx) {
  __shared__ __align__(16) bf16 Ks[KV_TILE * HEAD];
  __shared__ __align__(16) bf16 Vs[KV_TILE * HEAD];
  __shared__ __align__(16) bf16 Ps[4][16 * KV_TILE];

  const int h = blockIdx.y;
  const int b = blockIdx.z;
  const int tid = threadIdx.x;
  const int lane = tid & 31;
  const int wave = tid >> 5;
  const int qBase = blockIdx.x * 64 + wave * 16;
  const int half = (lane >> 4) & 1;
  const int nl = lane & 15;

  const size_t headOff = ((size_t)b * NH + h) * SEQ;

  // Q strip 16x64 held in registers as two A-fragments.
  const bf16* qsrc = Q + (headOff + qBase) * HEAD;
  v16bf qa0 = load_frag_rm(qsrc, HEAD, 0, lane);
  v16bf qa1 = load_frag_rm(qsrc, HEAD, 32, lane);

  v8f o[4] = {};
  float rmax[8], rsum[8];
  #pragma unroll
  for (int r = 0; r < 8; ++r) { rmax[r] = -1e30f; rsum[r] = 0.f; }

  for (int kv0 = 0; kv0 < SEQ; kv0 += KV_TILE) {
    __syncthreads();
    const bf16* kgp = Kt + (headOff + kv0) * HEAD;
    const bf16* vgp = Vt + (headOff + kv0) * HEAD;
    // 64x64 bf16 tiles: 512 x 16B units each, 4 per thread (async -> LDS).
    #pragma unroll
    for (int i = 0; i < 4; ++i) {
      int u = tid + i * ATT_THREADS;
#if HAVE_ASYNC_LDS
      async_copy16(kgp + (size_t)u * 8, Ks + (size_t)u * 8);
      async_copy16(vgp + (size_t)u * 8, Vs + (size_t)u * 8);
#else
      ((uint4*)Ks)[u] = ((const uint4*)kgp)[u];
      ((uint4*)Vs)[u] = ((const uint4*)vgp)[u];
#endif
    }
    if (kv0 + KV_TILE < SEQ) {
      __builtin_prefetch(kgp + (size_t)KV_TILE * HEAD, 0, 1);
      __builtin_prefetch(vgp + (size_t)KV_TILE * HEAD, 0, 1);
    }
#if HAVE_ASYNC_LDS
    wait_async0();
#endif
    __syncthreads();

    // Scores S = Q(16x64) . K^T -> four 16x16 tiles.
    v8f s[4];
    #pragma unroll
    for (int t = 0; t < 4; ++t) {
      v8f acc = {};
      acc = wmma_bf16(qa0, load_frag_rm(&Ks[(t * 16) * HEAD], HEAD, 0, lane), acc);
      acc = wmma_bf16(qa1, load_frag_rm(&Ks[(t * 16) * HEAD], HEAD, 32, lane), acc);
      s[t] = acc;
    }

    // scale + mask + running row max (reduce across 16-lane half-waves;
    // xor masks < 16 keep the two halves independent, matching C layout).
    float rmax_new[8];
    #pragma unroll
    for (int r = 0; r < 8; ++r) rmax_new[r] = rmax[r];
    #pragma unroll
    for (int t = 0; t < 4; ++t) {
      #pragma unroll
      for (int r = 0; r < 8; ++r) {
        int qg = qBase + r + half * 8;
        int kg = kv0 + t * 16 + nl;
        float mv = mask[((size_t)b * SEQ + qg) * SEQ + kg];
        float sv = s[t][r] * 0.125f + (1.0f - mv) * (-1e9f);
        s[t][r] = sv;
        float mx = sv;
        mx = fmaxf(mx, __shfl_xor(mx, 1, 32));
        mx = fmaxf(mx, __shfl_xor(mx, 2, 32));
        mx = fmaxf(mx, __shfl_xor(mx, 4, 32));
        mx = fmaxf(mx, __shfl_xor(mx, 8, 32));
        rmax_new[r] = fmaxf(rmax_new[r], mx);
      }
    }

    // Rescale running sum / output.
    #pragma unroll
    for (int r = 0; r < 8; ++r) {
      float f = __expf(rmax[r] - rmax_new[r]);
      rsum[r] *= f;
      #pragma unroll
      for (int t = 0; t < 4; ++t) o[t][r] *= f;
      rmax[r] = rmax_new[r];
    }

    // P = exp(S - max); accumulate row sums; spill P (bf16) to per-wave LDS.
    #pragma unroll
    for (int t = 0; t < 4; ++t) {
      #pragma unroll
      for (int r = 0; r < 8; ++r) {
        float p = __expf(s[t][r] - rmax[r]);
        float sm = p;
        sm += __shfl_xor(sm, 1, 32);
        sm += __shfl_xor(sm, 2, 32);
        sm += __shfl_xor(sm, 4, 32);
        sm += __shfl_xor(sm, 8, 32);
        rsum[r] += sm;
        Ps[wave][(r + half * 8) * KV_TILE + t * 16 + nl] = (bf16)p;
      }
    }

    // O += P(16x64) @ V(64x64). Same-wave LDS ops are in-order: no barrier.
    #pragma unroll
    for (int kk = 0; kk < 2; ++kk) {
      v16bf pa = load_frag_rm(&Ps[wave][0], KV_TILE, kk * 32, lane);
      #pragma unroll
      for (int t = 0; t < 4; ++t) {
        v16bf bv2 = load_frag_kn(Vs, HEAD, kk * 32, t * 16, lane);
        o[t] = wmma_bf16(pa, bv2, o[t]);
      }
    }
  }

  // Normalize, stage in the per-wave P buffer, coalesced b128 stores
  // into ctx [B][S][D] at col = h*HEAD + d (head-major concat).
  #pragma unroll
  for (int t = 0; t < 4; ++t)
    #pragma unroll
    for (int r = 0; r < 8; ++r)
      Ps[wave][(r + half * 8) * KV_TILE + t * 16 + nl] =
          (bf16)(o[t][r] / rsum[r]);
  #pragma unroll
  for (int i = 0; i < 4; ++i) {
    int u = lane + i * 32;            // 0..127 (16 rows x 8 x uint4)
    int r = u >> 3, c8 = (u & 7) * 8;
    *(uint4*)&ctx[((size_t)b * SEQ + qBase + r) * D_MODEL + h * HEAD + c8] =
        *(const uint4*)&Ps[wave][r * KV_TILE + c8];
  }
}

// ---- Kernel 4: output projection (fp32 out + bias) -------------------------

__global__ __launch_bounds__(GEMM_THREADS)
void out_proj_kernel(const bf16* __restrict__ ctx, const bf16* __restrict__ WoB,
                     const float* __restrict__ bo, float* __restrict__ out) {
  __shared__ __align__(16) bf16 As[TILE_M][TILE_K];
  __shared__ __align__(16) bf16 Bs[TILE_N][TILE_K];
  __shared__ __align__(16) float Cso[TILE_M][TILE_N];

  const int m0 = blockIdx.x * TILE_M;
  const int n0 = blockIdx.y * TILE_N;
  const int tid = threadIdx.x;
  const int lane = tid & 31;
  const int wave = tid >> 5;
  const int mOff = (wave >> 1) * 32;
  const int nOff = (wave & 1) * 32;

  v8f c[2][2] = {};

  for (int k0 = 0; k0 < D_MODEL; k0 += TILE_K) {
    // Both tiles are bf16 in memory: 256 x 16B units each, async -> LDS.
    #pragma unroll
    for (int i = 0; i < 2; ++i) {
      int u = tid + i * GEMM_THREADS;
      int r = u >> 2, seg = (u & 3) * 8;
      const bf16* ga = &ctx[(size_t)(m0 + r) * D_MODEL + k0 + seg];
      const bf16* gb = &WoB[(size_t)(n0 + r) * D_MODEL + k0 + seg];
#if HAVE_ASYNC_LDS
      async_copy16(ga, &As[r][seg]);
      async_copy16(gb, &Bs[r][seg]);
#else
      *(uint4*)&As[r][seg] = *(const uint4*)ga;
      *(uint4*)&Bs[r][seg] = *(const uint4*)gb;
#endif
    }
#if HAVE_ASYNC_LDS
    wait_async0();
#endif
    __syncthreads();
    #pragma unroll
    for (int mi = 0; mi < 2; ++mi) {
      v16bf a = load_frag_rm(&As[mOff + mi * 16][0], TILE_K, 0, lane);
      #pragma unroll
      for (int ni = 0; ni < 2; ++ni) {
        v16bf b = load_frag_rm(&Bs[nOff + ni * 16][0], TILE_K, 0, lane);
        c[mi][ni] = wmma_bf16(a, b, c[mi][ni]);
      }
    }
    __syncthreads();
  }

  // Epilogue: +bias, stage fp32 tile in LDS, coalesced float4 stores.
  const int half = (lane >> 4) & 1;
  const int nl = lane & 15;
  #pragma unroll
  for (int mi = 0; mi < 2; ++mi)
    #pragma unroll
    for (int ni = 0; ni < 2; ++ni) {
      float bv = bo[n0 + nOff + ni * 16 + nl];
      #pragma unroll
      for (int r = 0; r < 8; ++r)
        Cso[mOff + mi * 16 + r + half * 8][nOff + ni * 16 + nl] =
            c[mi][ni][r] + bv;
    }
  __syncthreads();
  #pragma unroll
  for (int i = 0; i < 8; ++i) {
    int u = tid + i * GEMM_THREADS;   // 0..1023 (64 rows x 16 x float4)
    int r = u >> 4, c4 = (u & 15) * 4;
    *(float4*)&out[(size_t)(m0 + r) * D_MODEL + n0 + c4] =
        *(const float4*)&Cso[r][c4];
  }
}

// ---- Launch ----------------------------------------------------------------

extern "C" void kernel_launch(void* const* d_in, const int* in_sizes, int n_in,
                              void* d_out, int out_size, void* d_ws, size_t ws_size,
                              hipStream_t stream) {
  const float* x    = (const float*)d_in[0];
  const float* y    = (const float*)d_in[1];
  const float* mask = (const float*)d_in[2];
  const float* Wq   = (const float*)d_in[3];
  const float* bq   = (const float*)d_in[4];
  const float* Wk   = (const float*)d_in[5];
  const float* bk   = (const float*)d_in[6];
  const float* Wv   = (const float*)d_in[7];
  const float* bv   = (const float*)d_in[8];
  const float* Wo   = (const float*)d_in[9];
  const float* bo   = (const float*)d_in[10];
  float* out = (float*)d_out;

  const size_t WELEMS = (size_t)D_MODEL * D_MODEL;  // 1M
  const size_t QKV    = (size_t)ROWS * D_MODEL;     // 4M
  bf16* WqB = (bf16*)d_ws;
  bf16* WkB = WqB + WELEMS;
  bf16* WvB = WkB + WELEMS;
  bf16* WoB = WvB + WELEMS;
  bf16* Qb  = WoB + WELEMS;
  bf16* Kb  = Qb + QKV;
  bf16* Vb  = Kb + QKV;
  bf16* Cb  = Vb + QKV;   // total 40 MB of workspace

  dim3 pk((unsigned)((WELEMS / 4 + 255) / 256));
  pack_w_kernel<<<pk, 256, 0, stream>>>(Wq, WqB, (int)WELEMS);
  pack_w_kernel<<<pk, 256, 0, stream>>>(Wk, WkB, (int)WELEMS);
  pack_w_kernel<<<pk, 256, 0, stream>>>(Wv, WvB, (int)WELEMS);
  pack_w_kernel<<<pk, 256, 0, stream>>>(Wo, WoB, (int)WELEMS);

  dim3 gq(ROWS / TILE_M, D_MODEL / TILE_N, 3);
  qkv_proj_kernel<<<gq, GEMM_THREADS, 0, stream>>>(x, y, WqB, WkB, WvB,
                                                   bq, bk, bv, Qb, Kb, Vb);

  dim3 ga(SEQ / 64, NH, BATCH);
  attention_kernel<<<ga, ATT_THREADS, 0, stream>>>(Qb, Kb, Vb, mask, Cb);

  dim3 go(ROWS / TILE_M, D_MODEL / TILE_N);
  out_proj_kernel<<<go, GEMM_THREADS, 0, stream>>>(Cb, WoB, bo, out);
}
